// KMeansLoss_3917010174520
// MI455X (gfx1250) — compile-verified, matrix operand order checked by reading, not run on hardware
//
#include <hip/hip_runtime.h>
#include <hip/hip_bf16.h>

typedef __attribute__((ext_vector_type(16))) __bf16 v16bf;
typedef __attribute__((ext_vector_type(8)))  float  v8f;

#define D_DIM 128
#define K_CENT 256
#define ROW_TILES_PER_WAVE 4
#define ROWS_PER_BLOCK (8 * ROW_TILES_PER_WAVE * 16)        // 512
// center-plane row stride: 68 dwords (136 halves) -> conflict-free b128 frag loads
#define CSTRIDE_DW 68
#define LDS_PLANE_BYTES (K_CENT * CSTRIDE_DW * 4)           // 69632
#define LDS_C2_OFF      (2 * LDS_PLANE_BYTES)               // 139264
#define LDS_TOTAL       (LDS_C2_OFF + K_CENT * 4)           // 140288

union U16BF { uint4 q[2]; v16bf v; };

__global__ void kmeans_zero_out(float* out) { out[0] = 0.0f; }

__global__ __launch_bounds__(256) void kmeans_min_dist(
    const float* __restrict__ feat, const float* __restrict__ cent,
    float* __restrict__ out, int N)
{
    extern __shared__ char smem[];
    unsigned short* ldsHi16 = (unsigned short*)smem;
    unsigned short* ldsLo16 = (unsigned short*)(smem + LDS_PLANE_BYTES);
    float* c2s = (float*)(smem + LDS_C2_OFF);

    const int tid  = threadIdx.x;
    const int wid  = tid >> 5;
    const int lane = tid & 31;
    const int g    = lane >> 4;   // K-half selector for A/B fragments
    const int nl   = lane & 15;

    // ---- Phase 1a: centers f32 -> bf16 hi/lo planes in LDS (coalesced) ----
    #pragma unroll
    for (int i = 0; i < (K_CENT * D_DIM) / 256; ++i) {
        int idx = tid + i * 256;
        int n = idx >> 7, k = idx & 127;
        float x = cent[idx];
        __bf16 h = (__bf16)x;
        float  r = x - (float)h;
        __bf16 l = (__bf16)r;
        ldsHi16[n * (2 * CSTRIDE_DW) + k] = __builtin_bit_cast(unsigned short, h);
        ldsLo16[n * (2 * CSTRIDE_DW) + k] = __builtin_bit_cast(unsigned short, l);
    }
    __syncthreads();

    // ---- Phase 1b: ||c||^2 from reconstructed centers ----
    if (tid < K_CENT) {
        float s = 0.f;
        for (int k = 0; k < D_DIM; ++k) {
            __bf16 h = __builtin_bit_cast(__bf16, ldsHi16[tid * (2 * CSTRIDE_DW) + k]);
            __bf16 l = __builtin_bit_cast(__bf16, ldsLo16[tid * (2 * CSTRIDE_DW) + k]);
            float x = (float)h + (float)l;
            s += x * x;
        }
        c2s[tid] = s;
    }
    __syncthreads();

    const uint4* hi4 = (const uint4*)ldsHi16;
    const uint4* lo4 = (const uint4*)ldsLo16;

    // B fragment chunk loader: 4 ds_load_b128 per call
    U16BF xh, xl, yh, yl;
    auto loadChunk = [&](int t, int c, U16BF& bh, U16BF& bl) {
        int bidx = (t * 16 + nl) * (CSTRIDE_DW / 4) + 4 * c + g;
        bh.q[0] = hi4[bidx]; bh.q[1] = hi4[bidx + 2];   // run2 = +8 dwords
        bl.q[0] = lo4[bidx]; bl.q[1] = lo4[bidx + 2];
    };

    v16bf ahi[4], alo[4];
    auto wmma3 = [&](int c, const U16BF& bh, const U16BF& bl, v8f acc) -> v8f {
        acc = __builtin_amdgcn_wmma_f32_16x16x32_bf16(
                  false, ahi[c], false, bh.v, (short)0, acc, false, false);
        acc = __builtin_amdgcn_wmma_f32_16x16x32_bf16(
                  false, alo[c], false, bh.v, (short)0, acc, false, false);
        acc = __builtin_amdgcn_wmma_f32_16x16x32_bf16(
                  false, ahi[c], false, bl.v, (short)0, acc, false, false);
        return acc;
    };

    // Prime the ping-pong pipeline with (tile 0, chunk 0); the t==15 tail
    // load below re-primes it for every subsequent row-tile iteration.
    loadChunk(0, 0, xh, xl);

    const int blockRow0 = blockIdx.x * ROWS_PER_BLOCK;
    float totalSum = 0.f;

    for (int rt = 0; rt < ROW_TILES_PER_WAVE; ++rt) {
        // ---- A fragments (features) + ||f||^2 ----
        // 16-bit A 16x32 layout: both lane halves hold row M=nl; half g owns
        // K = 8g+{0..7} and 16+8g+{0..7} of each 32-wide chunk.
        const int rowBase = blockRow0 + (wid * ROW_TILES_PER_WAVE + rt) * 16;
        int arow = rowBase + nl; if (arow >= N) arow = N - 1;
        const float4* fb = (const float4*)(feat + (size_t)arow * D_DIM);

        float sumsq = 0.f;
        #pragma unroll
        for (int c = 0; c < 4; ++c) {
            int q0 = (32 * c + 8 * g) >> 2;
            float4 a = fb[q0], b = fb[q0 + 1];
            float4 cc = fb[q0 + 4], d = fb[q0 + 5];
            float xs[16] = { a.x, a.y, a.z, a.w,  b.x, b.y, b.z, b.w,
                             cc.x, cc.y, cc.z, cc.w,  d.x, d.y, d.z, d.w };
            #pragma unroll
            for (int h = 0; h < 16; ++h) {
                float x = xs[h];
                sumsq += x * x;
                __bf16 hb = (__bf16)x;
                float  rr = x - (float)hb;
                ahi[c][h] = hb;
                alo[c][h] = (__bf16)rr;
            }
        }
        // prefetch next row-tile's feature rows while we crunch this one
        if (rt + 1 < ROW_TILES_PER_WAVE) {
            int nrow = rowBase + 16 + nl; if (nrow >= N) nrow = N - 1;
            __builtin_prefetch(feat + (size_t)nrow * D_DIM + g * 64, 0, 3);
        }

        sumsq += __shfl_xor(sumsq, 16, 32);           // combine K-halves
        float f2r[8];
        #pragma unroll
        for (int r = 0; r < 8; ++r)
            f2r[r] = __shfl(sumsq, r + 8 * g, 32);    // f2 in C-layout rows

        float minacc[8];
        #pragma unroll
        for (int r = 0; r < 8; ++r) minacc[r] = 3.4e38f;

        // ---- 16 center tiles, chunk-level double-buffered B fragments ----
        for (int t = 0; t < 16; ++t) {
            v8f acc = {};
            loadChunk(t, 1, yh, yl);
            acc = wmma3(0, xh, xl, acc);
            loadChunk(t, 2, xh, xl);
            acc = wmma3(1, yh, yl, acc);
            loadChunk(t, 3, yh, yl);
            acc = wmma3(2, xh, xl, acc);
            loadChunk((t + 1) & 15, 0, xh, xl);       // primes next tile / next rt
            acc = wmma3(3, yh, yl, acc);

            float c2v = c2s[t * 16 + nl];
            #pragma unroll
            for (int r = 0; r < 8; ++r) {
                float d2 = f2r[r] + c2v - 2.0f * acc[r];
                d2 = fmaxf(d2, 0.0f);
                minacc[r] = fminf(minacc[r], d2);
            }
        }

        // min across the 16 lanes of each wave half (all 256 centers covered)
        #pragma unroll
        for (int off = 1; off < 16; off <<= 1) {
            #pragma unroll
            for (int r = 0; r < 8; ++r)
                minacc[r] = fminf(minacc[r], __shfl_xor(minacc[r], off, 32));
        }

        if (nl == 0) {  // lanes 0 (rows 0-7) and 16 (rows 8-15)
            #pragma unroll
            for (int r = 0; r < 8; ++r) {
                int row = rowBase + r + 8 * g;
                if (row < N) totalSum += sqrtf(minacc[r]);
            }
        }
    }

    if (nl == 0 && totalSum != 0.f)
        atomicAdd(out, totalSum * (1.0f / (float)N));
}

extern "C" void kernel_launch(void* const* d_in, const int* in_sizes, int n_in,
                              void* d_out, int out_size, void* d_ws, size_t ws_size,
                              hipStream_t stream) {
    const float* feat = (const float*)d_in[0];
    const float* cent = (const float*)d_in[1];
    float* out = (float*)d_out;
    int N = in_sizes[0] / D_DIM;   // 262144
    kmeans_zero_out<<<1, 1, 0, stream>>>(out);
    int blocks = (N + ROWS_PER_BLOCK - 1) / ROWS_PER_BLOCK;
    kmeans_min_dist<<<blocks, 256, LDS_TOTAL, stream>>>(feat, cent, out, N);
}
